// PairEnergies_87514253623857
// MI455X (gfx1250) — matrix operand order repaired and checked by reading, not compiled, not run on hardware
//
#include <hip/hip_runtime.h>

typedef __bf16 bf16_t;
typedef __attribute__((ext_vector_type(16))) __bf16 v16bf;
typedef __attribute__((ext_vector_type(8)))  __bf16 v8bf;
typedef __attribute__((ext_vector_type(8)))  float   v8f;

#define B_     2
#define N_     1024
#define K_     30
#define H_     128
#define L_     3
#define ODIM_  400
#define SCALE_ 30.0f
#define ROWS_E (B_*N_*K_)   /* 61440 */
#define ROWS_V (B_*N_)      /* 2048  */

static inline int cdiv_i(int a, int b) { return (a + b - 1) / b; }

#ifndef __has_builtin
#define __has_builtin(x) 0
#endif

__device__ __forceinline__ void sched_fence() {
#if __has_builtin(__builtin_amdgcn_sched_barrier)
  __builtin_amdgcn_sched_barrier(0);  // nothing may cross this point
#endif
}

// ---------------------------------------------------------------------------
// WMMA helpers (gfx1250, wave32). D = A(16x32 bf16) * B(32x16 bf16) + C(f32)
// ---------------------------------------------------------------------------
__device__ __forceinline__ v8f wmma_bf16(v16bf a, v16bf b, v8f c) {
  return __builtin_amdgcn_wmma_f32_16x16x32_bf16(false, a, false, b, (short)0, c,
                                                 false, false);
}

// A fragment from LDS row-major bf16 tile (p = row base + kc*32 + koff).
__device__ __forceinline__ v16bf load_afrag(const bf16_t* p) {
  v8bf lo = *(const v8bf*)p;         // K = koff .. koff+7
  v8bf hi = *(const v8bf*)(p + 16);  // K = 16+koff .. 16+koff+7
  v16bf a;
#pragma unroll
  for (int i = 0; i < 8; ++i) { a[i] = lo[i]; a[i + 8] = hi[i]; }
  return a;
}

// B fragment from packed weights: each lane owns 16 contiguous halfs.
__device__ __forceinline__ v16bf load_bfrag(const bf16_t* p) {
  v8bf lo = *(const v8bf*)p;
  v8bf hi = *(const v8bf*)(p + 8);
  v16bf b;
#pragma unroll
  for (int i = 0; i < 8; ++i) { b[i] = lo[i]; b[i + 8] = hi[i]; }
  return b;
}

// Dual-M-subtile K-loop: one B fragment feeds two independent WMMA chains
// (rows m..m+15 and m+16..m+31). B fragments are fetched in groups of 4 into
// distinct live registers (so RA cannot recycle one destination and create
// WAR serialization), and the NEXT group's 8 b128 loads are issued before the
// current group's 8 WMMAs, pinned by a scheduling barrier -> load latency is
// covered by matrix work instead of s_wait_loadcnt 0 per chunk.
template <int KC>
__device__ __forceinline__ void gemm_tile2(const bf16_t* __restrict__ A0,
                                           const bf16_t* __restrict__ A1,
                                           const bf16_t* __restrict__ Pt,
                                           v8f& accA, v8f& accB) {
  constexpr int G = 4;
  static_assert(KC % G == 0, "KC must be a multiple of 4");
  constexpr int NG = KC / G;
  v16bf B[G];
#pragma unroll
  for (int i = 0; i < G; ++i) B[i] = load_bfrag(Pt + i * 512);
#pragma unroll
  for (int g = 0; g < NG; ++g) {
    v16bf Bn[G];
    if (g + 1 < NG) {
#pragma unroll
      for (int i = 0; i < G; ++i)
        Bn[i] = load_bfrag(Pt + ((g + 1) * G + i) * 512);
    }
    sched_fence();  // keep next-group loads above this group's WMMAs
#pragma unroll
    for (int i = 0; i < G; ++i) {
      const int kc = g * G + i;
      accA = wmma_bf16(load_afrag(A0 + kc * 32), B[i], accA);
      accB = wmma_bf16(load_afrag(A1 + kc * 32), B[i], accB);
    }
    sched_fence();
    if (g + 1 < NG) {
#pragma unroll
      for (int i = 0; i < G; ++i) B[i] = Bn[i];
    }
  }
}

// ---------------------------------------------------------------------------
// Weight repack: f32 row-major [K][N] -> bf16 WMMA-B tiles.
// Tile T = nt*Kc + kc (K-chunks contiguous per N-tile for streaming).
// Within a tile: lane L (0..31), half j (0..15):
//   n = nt*16 + (L&15);  k = kc*32 + (L>>4)*8 + (j&7) + (j>>3)*16   (B-matrix ISA layout)
// ---------------------------------------------------------------------------
__global__ void pack_weight_kernel(const float* __restrict__ W,
                                   bf16_t* __restrict__ P, int Kdim, int Ndim) {
  int Kc = (Kdim + 31) >> 5;
  int Nt = (Ndim + 15) >> 4;
  int total = Kc * Nt * 512;
  for (int idx = blockIdx.x * blockDim.x + threadIdx.x; idx < total;
       idx += gridDim.x * blockDim.x) {
    int tile = idx >> 9;
    int within = idx & 511;
    int lane = within >> 4;
    int j = within & 15;
    int kc = tile % Kc;
    int nt = tile / Kc;
    int n = nt * 16 + (lane & 15);
    int k = kc * 32 + (lane >> 4) * 8 + (j & 7) + (j >> 3) * 16;
    float v = (k < Kdim && n < Ndim) ? W[(size_t)k * Ndim + n] : 0.0f;
    P[idx] = (bf16_t)v;
  }
}

// ---------------------------------------------------------------------------
// Generic GEMM, Kdim = 128: Out[M,Nout] = act(A[M,128] @ Wp + bias)
// 256 threads = 8 waves; block tile = 32 rows; wave covers N-tiles w, w+8, ...
// ---------------------------------------------------------------------------
__global__ void gemm128_kernel(const float* __restrict__ A,
                               const bf16_t* __restrict__ Wp,
                               const float* __restrict__ bias,
                               float* __restrict__ Out, int Nout, int relu) {
  __shared__ __attribute__((aligned(16))) bf16_t As[32 * 128];
  const int row0 = blockIdx.x * 32;
  const int tid = threadIdx.x;
  for (int i = tid; i < 32 * 128; i += 256) {
    int r = i >> 7, c = i & 127;
    As[i] = (bf16_t)A[(size_t)(row0 + r) * 128 + c];
  }
  __syncthreads();
  const int w = tid >> 5, lane = tid & 31;
  const int m_lo = lane & 15;
  const int koff = (lane >> 4) * 8;
  const int mb = 8 * (lane >> 4);
  const int Nt = (Nout + 15) >> 4;
  const bf16_t* A0 = As + m_lo * 128 + koff;
  const bf16_t* A1 = As + (16 + m_lo) * 128 + koff;
  for (int t = w; t < Nt; t += 8) {
    v8f accA = {}, accB = {};
    gemm_tile2<4>(A0, A1, Wp + (size_t)t * 4 * 512 + lane * 16, accA, accB);
    int n = t * 16 + m_lo;
    if (n < Nout) {
      float bv = bias ? bias[n] : 0.0f;
#pragma unroll
      for (int r = 0; r < 8; ++r) {
        float vA = accA[r] + bv;
        float vB = accB[r] + bv;
        if (relu) { vA = fmaxf(vA, 0.0f); vB = fmaxf(vB, 0.0f); }
        Out[(size_t)(row0 + mb + r) * Nout + n] = vA;
        Out[(size_t)(row0 + 16 + mb + r) * Nout + n] = vB;
      }
    }
  }
}

// mask_attend[b,n,k] = x_mask[b, E_idx[b,n,k]] * x_mask[b,n]
__global__ void mask_attend_kernel(const float* __restrict__ x_mask,
                                   const int* __restrict__ E_idx,
                                   float* __restrict__ matt) {
  int idx = blockIdx.x * blockDim.x + threadIdx.x;
  if (idx >= ROWS_E) return;
  int b = idx / (N_ * K_);
  int n = (idx / K_) % N_;
  matt[idx] = x_mask[b * N_ + E_idx[idx]] * x_mask[b * N_ + n];
}

// ---------------------------------------------------------------------------
// Fused message MLP: rows are flat (b,n,k). Builds concat row in LDS, then
// relu(.@W1+b1) -> relu(.@W2+b2) -> .@W3+b3 to global Msg.
// node (edge_mode=0): [h_i | h_nb | h_E];  edge (=1): [h_i | h_E | h_j]
// Block tile = 32 rows, 8 waves; wave w owns output columns [16w, 16w+16).
// ---------------------------------------------------------------------------
__global__ void message_kernel(const float* __restrict__ hV,
                               const float* __restrict__ hE,
                               const int* __restrict__ E_idx,
                               const bf16_t* __restrict__ W1p, const float* __restrict__ b1,
                               const bf16_t* __restrict__ W2p, const float* __restrict__ b2,
                               const bf16_t* __restrict__ W3p, const float* __restrict__ b3,
                               float* __restrict__ Msg, int edge_mode) {
  __shared__ __attribute__((aligned(16))) bf16_t As[32 * 384];
  __shared__ __attribute__((aligned(16))) bf16_t M1[32 * 128];
  __shared__ __attribute__((aligned(16))) bf16_t M2[32 * 128];
  const int row0 = blockIdx.x * 32;
  const int tid = threadIdx.x;
  for (int i = tid; i < 32 * 384; i += 256) {
    int r = i / 384, c = i - r * 384;
    int g = row0 + r;
    int b = g / (N_ * K_);
    int rem = g - b * (N_ * K_);
    int n = rem / K_;
    int seg = c >> 7, cc = c & 127;
    float v;
    if (seg == 0) {
      v = hV[((size_t)b * N_ + n) * H_ + cc];
    } else if ((seg == 1) != (edge_mode != 0)) {  // gathered neighbor features
      int j = E_idx[g];
      v = hV[((size_t)b * N_ + j) * H_ + cc];
    } else {                                      // edge features
      v = hE[(size_t)g * H_ + cc];
    }
    As[i] = (bf16_t)v;
  }
  __syncthreads();
  const int w = tid >> 5, lane = tid & 31;
  const int m_lo = lane & 15;
  const int koff = (lane >> 4) * 8;
  const int mb = 8 * (lane >> 4);
  const int n = w * 16 + m_lo;
  // GEMM1: [32,384] @ [384,128], relu
  {
    v8f accA = {}, accB = {};
    gemm_tile2<12>(As + m_lo * 384 + koff, As + (16 + m_lo) * 384 + koff,
                   W1p + (size_t)w * 12 * 512 + lane * 16, accA, accB);
    float bv = b1[n];
#pragma unroll
    for (int r = 0; r < 8; ++r) {
      M1[(mb + r) * 128 + n] = (bf16_t)fmaxf(accA[r] + bv, 0.0f);
      M1[(16 + mb + r) * 128 + n] = (bf16_t)fmaxf(accB[r] + bv, 0.0f);
    }
  }
  __syncthreads();
  // GEMM2: [32,128] @ [128,128], relu
  {
    v8f accA = {}, accB = {};
    gemm_tile2<4>(M1 + m_lo * 128 + koff, M1 + (16 + m_lo) * 128 + koff,
                  W2p + (size_t)w * 4 * 512 + lane * 16, accA, accB);
    float bv = b2[n];
#pragma unroll
    for (int r = 0; r < 8; ++r) {
      M2[(mb + r) * 128 + n] = (bf16_t)fmaxf(accA[r] + bv, 0.0f);
      M2[(16 + mb + r) * 128 + n] = (bf16_t)fmaxf(accB[r] + bv, 0.0f);
    }
  }
  __syncthreads();
  // GEMM3: [32,128] @ [128,128] -> global f32
  {
    v8f accA = {}, accB = {};
    gemm_tile2<4>(M2 + m_lo * 128 + koff, M2 + (16 + m_lo) * 128 + koff,
                  W3p + (size_t)w * 4 * 512 + lane * 16, accA, accB);
    float bv = b3[n];
#pragma unroll
    for (int r = 0; r < 8; ++r) {
      Msg[(size_t)(row0 + mb + r) * H_ + n] = accA[r] + bv;
      Msg[(size_t)(row0 + 16 + mb + r) * H_ + n] = accB[r] + bv;
    }
  }
}

// dh = sum_k Msg[b,n,k,:]*mask_attend[b,n,k] / SCALE;  hV = LN(hV + dh)
__global__ void node_aggregate_ln_kernel(const float* __restrict__ Msg,
                                         const float* __restrict__ matt,
                                         float* __restrict__ hV) {
  int bn = blockIdx.x;           // 0..ROWS_V-1
  int c = threadIdx.x;           // 0..127
  const float* mrow = Msg + (size_t)bn * K_ * H_;
  const float* mk = matt + (size_t)bn * K_;
  float s = 0.0f;
  for (int k = 0; k < K_; ++k) s += mrow[k * H_ + c] * mk[k];
  float x = hV[(size_t)bn * H_ + c] + s * (1.0f / SCALE_);
  __shared__ float buf[H_];
  buf[c] = x; __syncthreads();
  for (int o = 64; o > 0; o >>= 1) { if (c < o) buf[c] += buf[c + o]; __syncthreads(); }
  float mu = buf[0] * (1.0f / H_); __syncthreads();
  float d = x - mu;
  buf[c] = d * d; __syncthreads();
  for (int o = 64; o > 0; o >>= 1) { if (c < o) buf[c] += buf[c + o]; __syncthreads(); }
  float var = buf[0] * (1.0f / H_);
  hV[(size_t)bn * H_ + c] = d * rsqrtf(var + 1e-5f);
}

// X = LN(X + D), per 128-wide row
__global__ void residual_ln_kernel(float* __restrict__ X, const float* __restrict__ D) {
  int row = blockIdx.x;
  int c = threadIdx.x;
  float x = X[(size_t)row * H_ + c] + D[(size_t)row * H_ + c];
  __shared__ float buf[H_];
  buf[c] = x; __syncthreads();
  for (int o = 64; o > 0; o >>= 1) { if (c < o) buf[c] += buf[c + o]; __syncthreads(); }
  float mu = buf[0] * (1.0f / H_); __syncthreads();
  float d = x - mu;
  buf[c] = d * d; __syncthreads();
  for (int o = 64; o > 0; o >>= 1) { if (c < o) buf[c] += buf[c + o]; __syncthreads(); }
  float var = buf[0] * (1.0f / H_);
  X[(size_t)row * H_ + c] = d * rsqrtf(var + 1e-5f);
}

// ---------------------------------------------------------------------------
// Fused FFN + residual + LN + mask:
//   X = LN(X + relu(X@Wd1+bd1)@Wd2+bd2) * rowmask[row]
// Block tile = 32 rows.
// ---------------------------------------------------------------------------
__global__ void ffn_ln_kernel(float* __restrict__ X,
                              const bf16_t* __restrict__ Wd1p, const float* __restrict__ bd1,
                              const bf16_t* __restrict__ Wd2p, const float* __restrict__ bd2,
                              const float* __restrict__ rowmask) {
  __shared__ __attribute__((aligned(16))) bf16_t Xb[32 * 128];
  __shared__ float Xf[32 * 128];
  __shared__ __attribute__((aligned(16))) bf16_t T[32 * 512];
  __shared__ float Dh[32 * 128];
  __shared__ float red[256];
  const int row0 = blockIdx.x * 32;
  const int tid = threadIdx.x;
  for (int i = tid; i < 32 * 128; i += 256) {
    int r = i >> 7, c = i & 127;
    float v = X[(size_t)(row0 + r) * 128 + c];
    Xf[i] = v; Xb[i] = (bf16_t)v;
  }
  __syncthreads();
  const int w = tid >> 5, lane = tid & 31;
  const int m_lo = lane & 15;
  const int koff = (lane >> 4) * 8;
  const int mb = 8 * (lane >> 4);
  // GEMM1: 128 -> 512, relu, into LDS bf16 (wave covers 4 strided N-tiles)
#pragma unroll
  for (int s = 0; s < 4; ++s) {
    int t = w + 8 * s;
    v8f accA = {}, accB = {};
    gemm_tile2<4>(Xb + m_lo * 128 + koff, Xb + (16 + m_lo) * 128 + koff,
                  Wd1p + (size_t)t * 4 * 512 + lane * 16, accA, accB);
    int n = t * 16 + m_lo;
    float bv = bd1[n];
#pragma unroll
    for (int r = 0; r < 8; ++r) {
      T[(mb + r) * 512 + n] = (bf16_t)fmaxf(accA[r] + bv, 0.0f);
      T[(16 + mb + r) * 512 + n] = (bf16_t)fmaxf(accB[r] + bv, 0.0f);
    }
  }
  __syncthreads();
  // GEMM2: 512 -> 128, + bias + residual into Dh
  {
    v8f accA = {}, accB = {};
    gemm_tile2<16>(T + m_lo * 512 + koff, T + (16 + m_lo) * 512 + koff,
                   Wd2p + (size_t)w * 16 * 512 + lane * 16, accA, accB);
    int n = w * 16 + m_lo;
    float bv = bd2[n];
#pragma unroll
    for (int r = 0; r < 8; ++r) {
      Dh[(mb + r) * 128 + n] = accA[r] + bv + Xf[(mb + r) * 128 + n];
      Dh[(16 + mb + r) * 128 + n] = accB[r] + bv + Xf[(16 + mb + r) * 128 + n];
    }
  }
  __syncthreads();
  // LayerNorm + mask: 8 threads per row (32 rows x 8 = 256)
  int rr = tid >> 3, sub = tid & 7;
  float ps = 0.0f;
#pragma unroll
  for (int q = 0; q < 16; ++q) ps += Dh[rr * 128 + sub * 16 + q];
  red[tid] = ps; __syncthreads();
  for (int o = 4; o > 0; o >>= 1) { if (sub < o) red[tid] += red[tid + o]; __syncthreads(); }
  float mu = red[rr << 3] * (1.0f / H_); __syncthreads();
  float pv = 0.0f;
#pragma unroll
  for (int q = 0; q < 16; ++q) {
    float d = Dh[rr * 128 + sub * 16 + q] - mu;
    pv += d * d;
  }
  red[tid] = pv; __syncthreads();
  for (int o = 4; o > 0; o >>= 1) { if (sub < o) red[tid] += red[tid + o]; __syncthreads(); }
  float var = red[rr << 3] * (1.0f / H_);
  float rinv = rsqrtf(var + 1e-5f);
  float mk = rowmask[row0 + rr];
#pragma unroll
  for (int q = 0; q < 16; ++q) {
    int c = sub * 16 + q;
    X[(size_t)(row0 + rr) * 128 + c] = (Dh[rr * 128 + c] - mu) * rinv * mk;
  }
}

// out[b,i,:] = etab[b, inv_mapping[b,i], :]
__global__ void merge_kernel(const float* __restrict__ etab,
                             const int* __restrict__ inv,
                             float* __restrict__ out) {
  size_t total = (size_t)ROWS_E * ODIM_;
  for (size_t idx = (size_t)blockIdx.x * blockDim.x + threadIdx.x; idx < total;
       idx += (size_t)gridDim.x * blockDim.x) {
    size_t row = idx / ODIM_;
    int c = (int)(idx - row * ODIM_);
    int b = (int)(row / (N_ * K_));
    int i = (int)(row - (size_t)b * (N_ * K_));
    int src = inv[b * (N_ * K_) + i];
    out[idx] = etab[((size_t)b * (N_ * K_) + src) * ODIM_ + c];
  }
}

// ---------------------------------------------------------------------------
extern "C" void kernel_launch(void* const* d_in, const int* in_sizes, int n_in,
                              void* d_out, int out_size, void* d_ws, size_t ws_size,
                              hipStream_t stream) {
  const float* h_V0   = (const float*)d_in[0];
  const float* h_E0   = (const float*)d_in[1];
  const int*   E_idx  = (const int*)d_in[2];
  const int*   invmap = (const int*)d_in[3];
  const float* x_mask = (const float*)d_in[4];
  const float* Wv_w = (const float*)d_in[5];  const float* Wv_b = (const float*)d_in[6];
  const float* We_w = (const float*)d_in[7];  const float* We_b = (const float*)d_in[8];
  const float* nW1 = (const float*)d_in[9];   const float* nb1 = (const float*)d_in[10];
  const float* nW2 = (const float*)d_in[11];  const float* nb2 = (const float*)d_in[12];
  const float* nW3 = (const float*)d_in[13];  const float* nb3 = (const float*)d_in[14];
  const float* nWd1 = (const float*)d_in[15]; const float* nbd1 = (const float*)d_in[16];
  const float* nWd2 = (const float*)d_in[17]; const float* nbd2 = (const float*)d_in[18];
  const float* eW1 = (const float*)d_in[19];  const float* eb1 = (const float*)d_in[20];
  const float* eW2 = (const float*)d_in[21];  const float* eb2 = (const float*)d_in[22];
  const float* eW3 = (const float*)d_in[23];  const float* eb3 = (const float*)d_in[24];
  const float* eWd1 = (const float*)d_in[25]; const float* ebd1 = (const float*)d_in[26];
  const float* eWd2 = (const float*)d_in[27]; const float* ebd2 = (const float*)d_in[28];
  const float* Wout_w = (const float*)d_in[29]; const float* Wout_b = (const float*)d_in[30];
  float* out = (float*)d_out;

  char* ws = (char*)d_ws;
  size_t off = 0;
  auto take = [&](size_t bytes) -> char* {
    char* p = ws + off;
    off = (off + bytes + 255) & ~(size_t)255;
    return p;
  };
  auto packhalfs = [](int K, int N) -> size_t {
    return (size_t)cdiv_i(K, 32) * cdiv_i(N, 16) * 512;
  };

  float* hV   = (float*)take((size_t)ROWS_V * H_ * 4);
  float* hE   = (float*)take((size_t)ROWS_E * H_ * 4);
  float* msg  = (float*)take((size_t)ROWS_E * H_ * 4);
  float* matt = (float*)take((size_t)ROWS_E * 4);
  float* etab = (float*)take((size_t)ROWS_E * ODIM_ * 4);

  bf16_t* pWv   = (bf16_t*)take(packhalfs(H_, H_) * 2);
  bf16_t* pWe   = (bf16_t*)take(packhalfs(H_, H_) * 2);
  bf16_t* pWout = (bf16_t*)take(packhalfs(H_, ODIM_) * 2);
  bf16_t *pnW1[L_], *pnW2[L_], *pnW3[L_], *pnWd1[L_], *pnWd2[L_];
  bf16_t *peW1[L_], *peW2[L_], *peW3[L_], *peWd1[L_], *peWd2[L_];
  for (int l = 0; l < L_; ++l) {
    pnW1[l]  = (bf16_t*)take(packhalfs(3 * H_, H_) * 2);
    pnW2[l]  = (bf16_t*)take(packhalfs(H_, H_) * 2);
    pnW3[l]  = (bf16_t*)take(packhalfs(H_, H_) * 2);
    pnWd1[l] = (bf16_t*)take(packhalfs(H_, 4 * H_) * 2);
    pnWd2[l] = (bf16_t*)take(packhalfs(4 * H_, H_) * 2);
    peW1[l]  = (bf16_t*)take(packhalfs(3 * H_, H_) * 2);
    peW2[l]  = (bf16_t*)take(packhalfs(H_, H_) * 2);
    peW3[l]  = (bf16_t*)take(packhalfs(H_, H_) * 2);
    peWd1[l] = (bf16_t*)take(packhalfs(H_, 4 * H_) * 2);
    peWd2[l] = (bf16_t*)take(packhalfs(4 * H_, H_) * 2);
  }
  if (off > ws_size) return;  // workspace too small: bail deterministically

  // ---- pack all weights to bf16 WMMA layout ----
  struct PD { const float* src; int K; int N; bf16_t* dst; };
  PD pds[33];
  int np = 0;
  pds[np++] = {Wv_w, H_, H_, pWv};
  pds[np++] = {We_w, H_, H_, pWe};
  pds[np++] = {Wout_w, H_, ODIM_, pWout};
  for (int l = 0; l < L_; ++l) {
    pds[np++] = {nW1 + (size_t)l * 3 * H_ * H_, 3 * H_, H_, pnW1[l]};
    pds[np++] = {nW2 + (size_t)l * H_ * H_, H_, H_, pnW2[l]};
    pds[np++] = {nW3 + (size_t)l * H_ * H_, H_, H_, pnW3[l]};
    pds[np++] = {nWd1 + (size_t)l * H_ * 4 * H_, H_, 4 * H_, pnWd1[l]};
    pds[np++] = {nWd2 + (size_t)l * 4 * H_ * H_, 4 * H_, H_, pnWd2[l]};
    pds[np++] = {eW1 + (size_t)l * 3 * H_ * H_, 3 * H_, H_, peW1[l]};
    pds[np++] = {eW2 + (size_t)l * H_ * H_, H_, H_, peW2[l]};
    pds[np++] = {eW3 + (size_t)l * H_ * H_, H_, H_, peW3[l]};
    pds[np++] = {eWd1 + (size_t)l * H_ * 4 * H_, H_, 4 * H_, peWd1[l]};
    pds[np++] = {eWd2 + (size_t)l * 4 * H_ * H_, 4 * H_, H_, peWd2[l]};
  }
  for (int i = 0; i < np; ++i) {
    int total = (int)packhalfs(pds[i].K, pds[i].N);
    pack_weight_kernel<<<cdiv_i(total, 256), 256, 0, stream>>>(
        pds[i].src, pds[i].dst, pds[i].K, pds[i].N);
  }

  // ---- embeddings ----
  gemm128_kernel<<<ROWS_V / 32, 256, 0, stream>>>(h_V0, pWv, Wv_b, hV, H_, 0);
  gemm128_kernel<<<ROWS_E / 32, 256, 0, stream>>>(h_E0, pWe, We_b, hE, H_, 0);
  mask_attend_kernel<<<cdiv_i(ROWS_E, 256), 256, 0, stream>>>(x_mask, E_idx, matt);

  // ---- L message-passing layers ----
  for (int l = 0; l < L_; ++l) {
    // node update
    message_kernel<<<ROWS_E / 32, 256, 0, stream>>>(
        hV, hE, E_idx, pnW1[l], nb1 + l * H_, pnW2[l], nb2 + l * H_,
        pnW3[l], nb3 + l * H_, msg, 0);
    node_aggregate_ln_kernel<<<ROWS_V, H_, 0, stream>>>(msg, matt, hV);
    ffn_ln_kernel<<<ROWS_V / 32, 256, 0, stream>>>(
        hV, pnWd1[l], nbd1 + l * 4 * H_, pnWd2[l], nbd2 + l * H_, x_mask);
    // edge update
    message_kernel<<<ROWS_E / 32, 256, 0, stream>>>(
        hV, hE, E_idx, peW1[l], eb1 + l * H_, peW2[l], eb2 + l * H_,
        peW3[l], eb3 + l * H_, msg, 1);
    residual_ln_kernel<<<ROWS_E, H_, 0, stream>>>(hE, msg);
    ffn_ln_kernel<<<ROWS_E / 32, 256, 0, stream>>>(
        hE, peWd1[l], ebd1 + l * 4 * H_, peWd2[l], ebd2 + l * H_, matt);
  }

  // ---- output projection + merge_dups ----
  gemm128_kernel<<<ROWS_E / 32, 256, 0, stream>>>(hE, pWout, Wout_b, etab, ODIM_, 0);
  merge_kernel<<<4096, 256, 0, stream>>>(etab, invmap, out);
}
